// StrassenAttention_71116068487393
// MI455X (gfx1250) — matrix operand order repaired and checked by reading, not compiled, not run on hardware
//
#include <hip/hip_runtime.h>
#include <hip/hip_bf16.h>

typedef _Float16 half_t;
typedef __attribute__((ext_vector_type(16))) _Float16 v16h;
typedef __attribute__((ext_vector_type(8)))  float    v8f;
typedef __attribute__((address_space(3))) half_t lds_half_t;

#define B_  4
#define N_  4096
#define C_  1024
#define H_  16
#define D_  64
#define M_  (B_ * N_)          // 16384 rows
#define LN_EPS 1e-5f

// ---------------------------------------------------------------------------
// helpers
// ---------------------------------------------------------------------------
union FragCvt { uint4 u[2]; v16h h; };
union U4H     { uint4 v;    half_t h[8]; };

__device__ __forceinline__ v16h ld_frag(const half_t* p0, const half_t* p1) {
    FragCvt c;
    c.u[0] = *reinterpret_cast<const uint4*>(p0);
    c.u[1] = *reinterpret_cast<const uint4*>(p1);
    return c.h;
}

__device__ __forceinline__ v8f wmma16x16x32(v16h a, v16h b, v8f c) {
    return __builtin_amdgcn_wmma_f32_16x16x32_f16(
        /*neg_a=*/false, a, /*neg_b=*/false, b,
        /*c_mod=*/(short)0, c, /*reuse_a=*/false, /*reuse_b=*/false);
}

// LDS byte-address (addrspace(3) offset) of a __shared__ pointer
__device__ __forceinline__ uint32_t lds_addr(const half_t* p) {
    return (uint32_t)(uintptr_t)(const lds_half_t*)p;
}

// per-lane async copy of 16 bytes global -> LDS (ASYNCcnt-tracked, no VGPR data)
__device__ __forceinline__ void async_copy_b128(uint32_t lds_off, const half_t* gptr) {
    asm volatile("global_load_async_to_lds_b128 %0, %1, off"
                 :: "v"(lds_off), "v"(gptr) : "memory");
}

__device__ __forceinline__ void wait_async_le4() {
    asm volatile("s_wait_asynccnt 0x4" ::: "memory");
}
__device__ __forceinline__ void wait_async_0() {
    asm volatile("s_wait_asynccnt 0x0" ::: "memory");
}

// ---------------------------------------------------------------------------
// f32 -> f16 convert (grid-stride)
// ---------------------------------------------------------------------------
__global__ __launch_bounds__(256) void k_cvt_f16(const float* __restrict__ in,
                                                 half_t* __restrict__ out, int n) {
    for (int i = blockIdx.x * blockDim.x + threadIdx.x; i < n;
         i += gridDim.x * blockDim.x)
        out[i] = (half_t)in[i];
}

// ---------------------------------------------------------------------------
// Tiled WMMA GEMM:  out[m,n] = sum_k A16[m,k]*W16[n,k] (+resid[m,n]) (+bias[n])
// block = 128x128 tile, 256 threads = 8 waves, wave tile 64x32 (4x2 subtiles)
// Double-buffered K-tiles staged with GLOBAL_LOAD_ASYNC_TO_LDS_B128;
// compute on buf[cur] overlaps the in-flight async copies of buf[nxt]
// (s_wait_asynccnt 4 retires only the previous tile's 4 per-wave ops).
// LDS rows padded to 40 halves (80B, 16B aligned) for b128 fragment loads.
// ---------------------------------------------------------------------------
__global__ __launch_bounds__(256) void k_gemm_f16(
    const half_t* __restrict__ A, const half_t* __restrict__ W,
    const float* __restrict__ resid, const float* __restrict__ bias,
    half_t* __restrict__ outh, float* __restrict__ outf,
    int M, int N, int K)
{
    __shared__ half_t As[2][128 * 40];
    __shared__ half_t Bs[2][128 * 40];

    const int t    = threadIdx.x;
    const int wave = t >> 5;
    const int lane = t & 31;
    const int lm   = lane & 15;
    const int lhalf= lane >> 4;
    const int m0   = blockIdx.x * 128;
    const int n0   = blockIdx.y * 128;
    const int wr   = wave >> 2;          // 0..1 : 64-row band
    const int wc   = wave & 3;           // 0..3 : 32-col band

    const int rowA = t >> 2, segA = t & 3;          // idx = t
    const int rowB = (t + 256) >> 2, segB = t & 3;  // idx = t + 256

    v8f acc[4][2];
    #pragma unroll
    for (int i = 0; i < 4; ++i)
        #pragma unroll
        for (int j = 0; j < 2; ++j) acc[i][j] = v8f{};

    // issue the 4 per-thread async copies for one K-tile into buffer `buf`
    auto issue_tile = [&](int k0, int buf) {
        async_copy_b128(lds_addr(&As[buf][rowA * 40 + segA * 8]),
                        &A[(size_t)(m0 + rowA) * K + k0 + segA * 8]);
        async_copy_b128(lds_addr(&Bs[buf][rowA * 40 + segA * 8]),
                        &W[(size_t)(n0 + rowA) * K + k0 + segA * 8]);
        async_copy_b128(lds_addr(&As[buf][rowB * 40 + segB * 8]),
                        &A[(size_t)(m0 + rowB) * K + k0 + segB * 8]);
        async_copy_b128(lds_addr(&Bs[buf][rowB * 40 + segB * 8]),
                        &W[(size_t)(n0 + rowB) * K + k0 + segB * 8]);
    };

    const int nk = K >> 5;               // K / 32
    issue_tile(0, 0);

    for (int kt = 0; kt < nk; ++kt) {
        const int cur = kt & 1;
        if (kt + 1 < nk) {
            issue_tile((kt + 1) << 5, cur ^ 1);
            if (kt + 2 < nk)             // warm L2 two tiles ahead
                __builtin_prefetch(&A[(size_t)(m0 + rowA) * K + ((kt + 2) << 5)], 0, 1);
            wait_async_le4();            // previous tile (buf[cur]) landed
        } else {
            wait_async_0();
        }
        __syncthreads();

        // ---- fragments from buf[cur]
        const int kbA = lhalf * 8;                  // A: K 0..7 / 8..15 (+16)
        v16h af[4];
        #pragma unroll
        for (int i = 0; i < 4; ++i) {
            const half_t* base = &As[cur][(wr * 64 + i * 16 + lm) * 40];
            af[i] = ld_frag(base + kbA, base + kbA + 16);
        }
        const int kbB = lhalf * 8;                  // B: K 0..7 / 8..15 (+16)
        v16h bf[2];
        #pragma unroll
        for (int j = 0; j < 2; ++j) {
            const half_t* base = &Bs[cur][(wc * 32 + j * 16 + lm) * 40];
            bf[j] = ld_frag(base + kbB, base + kbB + 16);
        }
        #pragma unroll
        for (int i = 0; i < 4; ++i)
            #pragma unroll
            for (int j = 0; j < 2; ++j)
                acc[i][j] = wmma16x16x32(af[i], bf[j], acc[i][j]);
        __syncthreads();                 // buf[cur] free for overwrite next iter
    }

    // ---- epilogue
    #pragma unroll
    for (int i = 0; i < 4; ++i) {
        #pragma unroll
        for (int j = 0; j < 2; ++j) {
            int mm0 = m0 + wr * 64 + i * 16 + (lhalf ? 8 : 0);
            int nn  = n0 + wc * 32 + j * 16 + lm;
            float bv = bias ? bias[nn] : 0.0f;
            #pragma unroll
            for (int r = 0; r < 8; ++r) {
                int m = mm0 + r;
                size_t off = (size_t)m * N + nn;
                float v = acc[i][j][r] + bv;
                if (resid) v += resid[off];
                if (outh) outh[off] = (half_t)v;
                else      outf[off] = v;
            }
        }
    }
}

// ---------------------------------------------------------------------------
// Stage 2: per (b,h) compute S1 = k1^T v1 / N, S2 = k2^T v2 / N  (64x64 f32)
// K/V tiles transposed on LDS store so fragment loads are contiguous b128.
// ---------------------------------------------------------------------------
__global__ __launch_bounds__(256) void k_stage2_S(
    const half_t* __restrict__ k1, const half_t* __restrict__ k2,
    const half_t* __restrict__ v1, const half_t* __restrict__ v2,
    float* __restrict__ Sg)
{
    __shared__ half_t kT1[64 * 40], kT2[64 * 40], vT1[64 * 40], vT2[64 * 40];

    const int bh = blockIdx.x;
    const int b  = bh >> 4, h = bh & 15;
    const int col0 = h * D_;
    const size_t rowbase = (size_t)b * N_;

    const int t    = threadIdx.x;
    const int wave = t >> 5;
    const int lane = t & 31;
    const int lm   = lane & 15;
    const int lhalf= lane >> 4;
    const int di   = wave >> 1;
    const int ei0  = (wave & 1) * 2;

    v8f acc[2][2];
    #pragma unroll
    for (int m = 0; m < 2; ++m)
        #pragma unroll
        for (int j = 0; j < 2; ++j) acc[m][j] = v8f{};

    const half_t* srcs[4] = {k1, k2, v1, v2};
    half_t* dsts[4] = {kT1, kT2, vT1, vT2};

    for (int n0 = 0; n0 < N_; n0 += 32) {
        // load 32x64 tile of each matrix, store transposed [d][n] into LDS
        #pragma unroll
        for (int m = 0; m < 4; ++m) {
            int row = t >> 3, dseg = t & 7;     // 32 rows x 8 segs
            U4H u;
            u.v = *reinterpret_cast<const uint4*>(
                &srcs[m][(rowbase + n0 + row) * C_ + col0 + dseg * 8]);
            #pragma unroll
            for (int e = 0; e < 8; ++e)
                dsts[m][(dseg * 8 + e) * 40 + row] = u.h[e];
        }
        __syncthreads();

        const int kbA = lhalf * 8;
        const half_t* ab1 = &kT1[(di * 16 + lm) * 40];
        const half_t* ab2 = &kT2[(di * 16 + lm) * 40];
        v16h a1 = ld_frag(ab1 + kbA, ab1 + kbA + 16);
        v16h a2 = ld_frag(ab2 + kbA, ab2 + kbA + 16);

        const int kbB = lhalf * 8;
        #pragma unroll
        for (int j = 0; j < 2; ++j) {
            const half_t* b1 = &vT1[((ei0 + j) * 16 + lm) * 40];
            const half_t* b2 = &vT2[((ei0 + j) * 16 + lm) * 40];
            v16h bv1 = ld_frag(b1 + kbB, b1 + kbB + 16);
            v16h bv2 = ld_frag(b2 + kbB, b2 + kbB + 16);
            acc[0][j] = wmma16x16x32(a1, bv1, acc[0][j]);
            acc[1][j] = wmma16x16x32(a2, bv2, acc[1][j]);
        }
        __syncthreads();
    }

    const float invN = 1.0f / (float)N_;
    #pragma unroll
    for (int m = 0; m < 2; ++m)
        #pragma unroll
        for (int j = 0; j < 2; ++j) {
            int d0 = di * 16 + (lhalf ? 8 : 0);
            int e  = (ei0 + j) * 16 + lm;
            #pragma unroll
            for (int r = 0; r < 8; ++r)
                Sg[(((size_t)bh * 2 + m) * 64 + d0 + r) * 64 + e] =
                    acc[m][j][r] * invN;
        }
}

// ---------------------------------------------------------------------------
// Stage 2b: v-means, fused T (stored TRANSPOSED [e][d], f16) and ybias
// T[d,e] = g1*S1*v2m[e] + g3*S2*v1m[e] + g4*S1*S2 ; ybias[e] = g2*sum_d S1*S2
// ---------------------------------------------------------------------------
__global__ __launch_bounds__(256) void k_stage2_T(
    const half_t* __restrict__ v1, const half_t* __restrict__ v2,
    const float* __restrict__ Sg,
    const float* __restrict__ g1, const float* __restrict__ g2,
    const float* __restrict__ g3, const float* __restrict__ g4,
    half_t* __restrict__ Tt, float* __restrict__ yb)
{
    __shared__ float red1[256], red2[256];
    __shared__ float vm1[64], vm2[64];

    const int bh = blockIdx.x;
    const int b  = bh >> 4, h = bh & 15;
    const int col0 = h * D_;
    const size_t rowbase = (size_t)b * N_;
    const int t = threadIdx.x;

    // ---- column means of v1, v2 over N
    {
        int e = t & 63, chunk = t >> 6;
        float s1 = 0.f, s2 = 0.f;
        for (int n = chunk * 1024; n < (chunk + 1) * 1024; ++n) {
            size_t off = (rowbase + n) * C_ + col0 + e;
            s1 += (float)v1[off];
            s2 += (float)v2[off];
        }
        red1[t] = s1; red2[t] = s2;
    }
    __syncthreads();
    if (t < 64) {
        float inv = 1.0f / (float)N_;
        vm1[t] = (red1[t] + red1[t + 64] + red1[t + 128] + red1[t + 192]) * inv;
        vm2[t] = (red2[t] + red2[t + 64] + red2[t + 128] + red2[t + 192]) * inv;
    }
    __syncthreads();

    const float G1 = g1[h], G2 = g2[h], G3 = g3[h], G4 = g4[h];
    const float* S1 = &Sg[((size_t)bh * 2 + 0) * 4096];
    const float* S2 = &Sg[((size_t)bh * 2 + 1) * 4096];

    for (int idx = t; idx < 4096; idx += 256) {
        int d = idx >> 6, e = idx & 63;
        float s1 = S1[d * 64 + e], s2 = S2[d * 64 + e];
        float tt = G1 * s1 * vm2[e] + G3 * s2 * vm1[e] + G4 * s1 * s2;
        Tt[(size_t)bh * 4096 + e * 64 + d] = (half_t)tt;   // transposed store
    }
    if (t < 64) {
        float s = 0.f;
        for (int d = 0; d < 64; ++d) s += S1[d * 64 + t] * S2[d * 64 + t];
        yb[bh * 64 + t] = G2 * s;
    }
}

// ---------------------------------------------------------------------------
// Stage 3: per (b,h), out[n, h*64+e] = (q @ T)[n,e] + ybias[e]    (f32)
// Q/T tiles staged with one-shot async-to-LDS copies.
// ---------------------------------------------------------------------------
__global__ __launch_bounds__(256) void k_stage3(
    const half_t* __restrict__ q, const half_t* __restrict__ Tt,
    const float* __restrict__ yb, float* __restrict__ outf)
{
    __shared__ half_t Qs[128 * 72];   // [m][k], stride 72 halves
    __shared__ half_t Ts[64 * 72];    // [e][k]

    const int mt = blockIdx.x;        // 0..31 m-tiles of 128
    const int bh = blockIdx.y;
    const int b  = bh >> 4, h = bh & 15;
    const int col0 = h * D_;
    const size_t rowbase = (size_t)b * N_;
    const int m0 = mt * 128;

    const int t    = threadIdx.x;
    const int wave = t >> 5;
    const int lane = t & 31;
    const int lm   = lane & 15;
    const int lhalf= lane >> 4;

    #pragma unroll
    for (int ii = 0; ii < 4; ++ii) {
        int idx = t + ii * 256;       // 1024 uint4
        int row = idx >> 3, seg = idx & 7;
        async_copy_b128(lds_addr(&Qs[row * 72 + seg * 8]),
                        &q[(rowbase + m0 + row) * C_ + col0 + seg * 8]);
    }
    #pragma unroll
    for (int ii = 0; ii < 2; ++ii) {
        int idx = t + ii * 256;       // 512 uint4
        int row = idx >> 3, seg = idx & 7;
        async_copy_b128(lds_addr(&Ts[row * 72 + seg * 8]),
                        &Tt[(size_t)bh * 4096 + row * 64 + seg * 8]);
    }
    wait_async_0();
    __syncthreads();

    v8f acc[4];
    #pragma unroll
    for (int j = 0; j < 4; ++j) acc[j] = v8f{};

    #pragma unroll
    for (int ks = 0; ks < 2; ++ks) {
        const int kbA = ks * 32 + lhalf * 8;
        const half_t* abase = &Qs[(wave * 16 + lm) * 72];
        v16h a = ld_frag(abase + kbA, abase + kbA + 16);
        const int kbB = ks * 32 + lhalf * 8;
        #pragma unroll
        for (int j = 0; j < 4; ++j) {
            const half_t* bbase = &Ts[(j * 16 + lm) * 72];
            v16h bb = ld_frag(bbase + kbB, bbase + kbB + 16);
            acc[j] = wmma16x16x32(a, bb, acc[j]);
        }
    }

    #pragma unroll
    for (int j = 0; j < 4; ++j) {
        int e = j * 16 + lm;
        float bias = yb[bh * 64 + e];
        int mm0 = m0 + wave * 16 + (lhalf ? 8 : 0);
        #pragma unroll
        for (int r = 0; r < 8; ++r)
            outf[(rowbase + mm0 + r) * C_ + col0 + e] = acc[j][r] + bias;
    }
}

// ---------------------------------------------------------------------------
// LayerNorm over C, f32 in -> f16 out
// ---------------------------------------------------------------------------
__global__ __launch_bounds__(256) void k_layernorm(
    const float* __restrict__ in, const float* __restrict__ gam,
    const float* __restrict__ bet, half_t* __restrict__ out)
{
    __shared__ float sA[256], sB[256];
    const int row = blockIdx.x;
    const int t = threadIdx.x;
    float v[4]; float s = 0.f, s2 = 0.f;
    #pragma unroll
    for (int i = 0; i < 4; ++i) {
        v[i] = in[(size_t)row * C_ + t + i * 256];
        s += v[i]; s2 += v[i] * v[i];
    }
    sA[t] = s; sB[t] = s2;
    __syncthreads();
    for (int stp = 128; stp > 0; stp >>= 1) {
        if (t < stp) { sA[t] += sA[t + stp]; sB[t] += sB[t + stp]; }
        __syncthreads();
    }
    const float mu  = sA[0] / (float)C_;
    const float var = sB[0] / (float)C_ - mu * mu;
    const float rs  = rsqrtf(var + LN_EPS);
    #pragma unroll
    for (int i = 0; i < 4; ++i) {
        int c = t + i * 256;
        out[(size_t)row * C_ + c] = (half_t)((v[i] - mu) * rs * gam[c] + bet[c]);
    }
}

// ---------------------------------------------------------------------------
// host launcher
// ---------------------------------------------------------------------------
extern "C" void kernel_launch(void* const* d_in, const int* in_sizes, int n_in,
                              void* d_out, int out_size, void* d_ws, size_t ws_size,
                              hipStream_t stream)
{
    const float* x    = (const float*)d_in[0];
    const float* Wq   = (const float*)d_in[1];  const float* bq  = (const float*)d_in[2];
    const float* Wk1  = (const float*)d_in[3];  const float* bk1 = (const float*)d_in[4];
    const float* Wk2  = (const float*)d_in[5];  const float* bk2 = (const float*)d_in[6];
    const float* Wv1  = (const float*)d_in[7];  const float* bv1 = (const float*)d_in[8];
    const float* Wv2  = (const float*)d_in[9];  const float* bv2 = (const float*)d_in[10];
    const float* g1   = (const float*)d_in[11]; const float* g2  = (const float*)d_in[12];
    const float* g3   = (const float*)d_in[13]; const float* g4  = (const float*)d_in[14];
    const float* ln_g = (const float*)d_in[15]; const float* ln_b= (const float*)d_in[16];
    const float* Wo   = (const float*)d_in[17]; const float* bo  = (const float*)d_in[18];
    float* outp = (float*)d_out;

    const size_t SZ_ACT = (size_t)M_ * C_ * sizeof(half_t);   // 32 MB
    const size_t SZ_W   = (size_t)C_ * C_ * sizeof(half_t);   //  2 MB
    char* ws = (char*)d_ws;
    size_t off = 0;
    half_t* xh  = (half_t*)(ws + off); off += SZ_ACT;   // also reused as outh
    half_t* qh  = (half_t*)(ws + off); off += SZ_ACT;
    half_t* k1h = (half_t*)(ws + off); off += SZ_ACT;
    half_t* k2h = (half_t*)(ws + off); off += SZ_ACT;
    half_t* v1h = (half_t*)(ws + off); off += SZ_ACT;
    half_t* v2h = (half_t*)(ws + off); off += SZ_ACT;
    half_t* Wh[6];
    for (int i = 0; i < 6; ++i) { Wh[i] = (half_t*)(ws + off); off += SZ_W; }
    float* Sg = (float*)(ws + off); off += (size_t)B_ * H_ * 2 * 64 * 64 * sizeof(float);
    half_t* Tt = (half_t*)(ws + off); off += (size_t)B_ * H_ * 64 * 64 * sizeof(half_t);
    float* yb  = (float*)(ws + off); off += (size_t)B_ * H_ * 64 * sizeof(float);
    half_t* outh = xh;   // alias: xh no longer needed after projections

    // 1. convert x and weights to f16
    k_cvt_f16<<<2048, 256, 0, stream>>>(x, xh, M_ * C_);
    const float* Wsrc[6] = {Wq, Wk1, Wk2, Wv1, Wv2, Wo};
    for (int i = 0; i < 6; ++i)
        k_cvt_f16<<<1024, 256, 0, stream>>>(Wsrc[i], Wh[i], C_ * C_);

    // 2. five residual projections: z = x + x W^T + b  (f16 out)
    dim3 gg(M_ / 128, C_ / 128);
    k_gemm_f16<<<gg, 256, 0, stream>>>(xh, Wh[0], x, bq,  qh,  nullptr, M_, C_, C_);
    k_gemm_f16<<<gg, 256, 0, stream>>>(xh, Wh[1], x, bk1, k1h, nullptr, M_, C_, C_);
    k_gemm_f16<<<gg, 256, 0, stream>>>(xh, Wh[2], x, bk2, k2h, nullptr, M_, C_, C_);
    k_gemm_f16<<<gg, 256, 0, stream>>>(xh, Wh[3], x, bv1, v1h, nullptr, M_, C_, C_);
    k_gemm_f16<<<gg, 256, 0, stream>>>(xh, Wh[4], x, bv2, v2h, nullptr, M_, C_, C_);

    // 3. S1, S2 per (b,h)
    k_stage2_S<<<B_ * H_, 256, 0, stream>>>(k1h, k2h, v1h, v2h, Sg);

    // 4. v-means + fused T (transposed) + ybias
    k_stage2_T<<<B_ * H_, 256, 0, stream>>>(v1h, v2h, Sg, g1, g2, g3, g4, Tt, yb);

    // 5. y = q @ T + ybias  -> pre-LN f32 parked in d_out
    k_stage3<<<dim3(N_ / 128, B_ * H_), 256, 0, stream>>>(qh, Tt, yb, outp);

    // 6. LayerNorm -> f16 (reuse xh buffer)
    k_layernorm<<<M_, 256, 0, stream>>>(outp, ln_g, ln_b, outh);

    // 7. final projection: out = LN(y) @ Wo^T + bo  (f32 -> d_out)
    k_gemm_f16<<<gg, 256, 0, stream>>>(outh, Wh[5], nullptr, bo, nullptr, outp,
                                       M_, C_, C_);

    (void)in_sizes; (void)n_in; (void)out_size; (void)ws_size;
}
